// MultiHeadAttention_39419209842854
// MI455X (gfx1250) — matrix operand order repaired and checked by reading, not compiled, not run on hardware
//
#include <hip/hip_runtime.h>
#include <hip/hip_bf16.h>

// ---------------------------------------------------------------------------
// MultiHeadAttention for MI455X (gfx1250, wave32, WMMA f16 -> f32 accum)
// B=2, S=2048, D=1024, H=16, DK=64
// ---------------------------------------------------------------------------

typedef _Float16 v16h __attribute__((ext_vector_type(16)));
typedef _Float16 h8   __attribute__((ext_vector_type(8)));
typedef _Float16 h4   __attribute__((ext_vector_type(4)));
typedef float    v8f  __attribute__((ext_vector_type(8)));

#define WMMA_F16(A, B, C) \
    __builtin_amdgcn_wmma_f32_16x16x32_f16(false, (A), false, (B), (short)0, (C), false, false)

constexpr int Bc  = 2;
constexpr int Sc  = 2048;
constexpr int Dc  = 1024;
constexpr int Hc  = 16;
constexpr int DKc = 64;
constexpr int NTOK = Bc * Sc;      // 4096 tokens

union AF { v16h v; h8 h[2]; };

__device__ __forceinline__ v8f zero8() {
    v8f z;
#pragma unroll
    for (int i = 0; i < 8; ++i) z[i] = 0.0f;
    return z;
}

// Async 16-byte global -> LDS copy (CDNA5 async path, ASYNCcnt tracked).
// LDS aperture keeps the byte offset in addr[31:0], so truncating the generic
// pointer yields the LDS address the instruction wants.
__device__ __forceinline__ void async_copy_b128(void* lds, const void* gptr) {
    unsigned int loff = (unsigned int)(unsigned long long)lds;
    asm volatile("global_load_async_to_lds_b128 %0, %1, off"
                 :: "v"(loff), "v"(gptr) : "memory");
}
__device__ __forceinline__ void wait_async0() {
    asm volatile("s_wait_asynccnt 0" ::: "memory");
}

// ---------------------------------------------------------------------------
// GEMM: out[m][n] = X[m][:] . W[n][:] + bias[n]
// MODE 0: X fp32, store f16 at [B,H,S,DK]      (Q, K projections)
// MODE 1: X fp32, store f16 at [B,H,DK,S]      (V projection, transposed)
// MODE 2: X f16 (attn buffer), store fp32 row-major [NTOK, D]  (output proj)
// Block tile 128x64, 256 threads = 8 waves, wave tile 32x32 (2x2 WMMA).
// MODE is a template parameter -> straight-line staging, no uniform branches.
// Steady-state loop always prefetches slice k+1 while slice k computes; the
// final slice is peeled.
// ---------------------------------------------------------------------------
template <int MODE>
__global__ __launch_bounds__(256) void proj_gemm_kernel(
    const float* __restrict__ X32, const _Float16* __restrict__ X16,
    const float* __restrict__ W,   const float* __restrict__ bias,
    _Float16* __restrict__ dst16,  float* __restrict__ dst32)
{
    constexpr int LDA = 40;                  // half-element stride (pad vs banks)
    __shared__ _Float16 As[128 * LDA];
    __shared__ _Float16 Bs[64 * LDA];

    const int t    = threadIdx.x;
    const int lane = t & 31;
    const int wave = t >> 5;
    const int wm   = wave & 3;               // 4 waves along M
    const int wn   = wave >> 2;              // 2 waves along N
    const int m0   = wm * 32;
    const int n0   = wn * 32;
    const int mBase = blockIdx.x * 128;
    const int nBase = blockIdx.y * 64;
    const int lm = lane & 15;
    const int hi = lane >> 4;

    v8f acc[2][2];
#pragma unroll
    for (int mi = 0; mi < 2; ++mi)
#pragma unroll
        for (int ni = 0; ni < 2; ++ni) acc[mi][ni] = zero8();

    float4 ra[4];      // A slice staging regs (fp32 modes)
    h8     ra16[2];    // A slice staging regs (f16 mode)
    float4 rb[2];      // B slice staging regs

    auto loadA = [&](int k0) {
        if (MODE != 2) {
#pragma unroll
            for (int i = 0; i < 4; ++i) {
                int lin = i * 1024 + t * 4;
                int r = lin >> 5, c = lin & 31;
                ra[i] = *reinterpret_cast<const float4*>(
                    X32 + (size_t)(mBase + r) * Dc + k0 + c);
            }
        } else {
#pragma unroll
            for (int i = 0; i < 2; ++i) {
                int u = i * 256 + t;             // h8 units: 4 per row
                int r = u >> 2, c8 = u & 3;
                ra16[i] = *reinterpret_cast<const h8*>(
                    X16 + (size_t)(mBase + r) * Dc + k0 + c8 * 8);
            }
        }
    };
    auto loadB = [&](int k0) {
#pragma unroll
        for (int i = 0; i < 2; ++i) {
            int lin = i * 1024 + t * 4;
            int r = lin >> 5, c = lin & 31;
            rb[i] = *reinterpret_cast<const float4*>(
                W + (size_t)(nBase + r) * Dc + k0 + c);
        }
    };
    auto commit = [&]() {
        if (MODE != 2) {
#pragma unroll
            for (int i = 0; i < 4; ++i) {
                int lin = i * 1024 + t * 4;
                int r = lin >> 5, c = lin & 31;
                h4 hh = { (_Float16)ra[i].x, (_Float16)ra[i].y,
                          (_Float16)ra[i].z, (_Float16)ra[i].w };
                *reinterpret_cast<h4*>(&As[r * LDA + c]) = hh;
            }
        } else {
#pragma unroll
            for (int i = 0; i < 2; ++i) {
                int u = i * 256 + t;
                int r = u >> 2, c8 = u & 3;
                *reinterpret_cast<h8*>(&As[r * LDA + c8 * 8]) = ra16[i];
            }
        }
#pragma unroll
        for (int i = 0; i < 2; ++i) {
            int lin = i * 1024 + t * 4;
            int r = lin >> 5, c = lin & 31;
            h4 hh = { (_Float16)rb[i].x, (_Float16)rb[i].y,
                      (_Float16)rb[i].z, (_Float16)rb[i].w };
            *reinterpret_cast<h4*>(&Bs[r * LDA + c]) = hh;
        }
    };
    auto compute = [&]() {
        AF a[2], b[2];
#pragma unroll
        for (int mi = 0; mi < 2; ++mi) {
            int row = m0 + mi * 16 + lm;
            a[mi].h[0] = *reinterpret_cast<const h8*>(&As[row * LDA + hi * 8]);
            a[mi].h[1] = *reinterpret_cast<const h8*>(&As[row * LDA + 16 + hi * 8]);
        }
#pragma unroll
        for (int ni = 0; ni < 2; ++ni) {
            int row = n0 + ni * 16 + lm;
            b[ni].h[0] = *reinterpret_cast<const h8*>(&Bs[row * LDA + hi * 8]);
            b[ni].h[1] = *reinterpret_cast<const h8*>(&Bs[row * LDA + 16 + hi * 8]);
        }
#pragma unroll
        for (int mi = 0; mi < 2; ++mi)
#pragma unroll
            for (int ni = 0; ni < 2; ++ni)
                acc[mi][ni] = WMMA_F16(a[mi].v, b[ni].v, acc[mi][ni]);
    };

    loadA(0);
    loadB(0);

    // steady state: always prefetch next slice while current slice computes
    for (int k0 = 0; k0 < Dc - 32; k0 += 32) {
        __syncthreads();
        commit();
        loadA(k0 + 32);
        loadB(k0 + 32);
        __syncthreads();
        compute();
    }
    // peeled last slice (no prefetch)
    __syncthreads();
    commit();
    __syncthreads();
    compute();

    // ---- epilogue: bias + layout-specific store ----
#pragma unroll
    for (int mi = 0; mi < 2; ++mi) {
#pragma unroll
        for (int ni = 0; ni < 2; ++ni) {
            int gn = nBase + n0 + ni * 16 + lm;
            float bval = bias[gn];
#pragma unroll
            for (int r = 0; r < 8; ++r) {
                int gm = mBase + m0 + mi * 16 + r + hi * 8;   // token index
                float val = acc[mi][ni][r] + bval;
                if (MODE == 2) {
                    dst32[(size_t)gm * Dc + gn] = val;
                } else {
                    int bb = gm >> 11, ss = gm & (Sc - 1);
                    int hh = gn >> 6,  dd = gn & (DKc - 1);
                    if (MODE == 0) {      // [B,H,S,DK]
                        dst16[(((size_t)(bb * Hc + hh)) * Sc + ss) * DKc + dd] =
                            (_Float16)val;
                    } else {              // [B,H,DK,S]  (V transposed)
                        dst16[(((size_t)(bb * Hc + hh)) * DKc + dd) * Sc + ss] =
                            (_Float16)val;
                    }
                }
            }
        }
    }
}

// ---------------------------------------------------------------------------
// Causal flash attention, one (b,h) per blockIdx.y, 64-query tile per block.
// 128 threads = 4 waves, each wave owns 16 query rows.
// K/V tiles are double-buffered in LDS and filled with async global->LDS
// copies (ASYNCcnt) so the DMA for block kb+1 overlaps the WMMAs of block kb.
// ---------------------------------------------------------------------------
__global__ __launch_bounds__(128) void flash_attn_kernel(
    const _Float16* __restrict__ Qh, const _Float16* __restrict__ Kh,
    const _Float16* __restrict__ Vt, _Float16* __restrict__ attnOut)
{
    constexpr int LD = 72;                   // padded half-stride
    __shared__ _Float16 Qs[64 * LD];
    __shared__ _Float16 Ks[2][64 * LD];
    __shared__ _Float16 Vs[2][64 * LD];
    __shared__ _Float16 Ps[64 * LD];

    const int t    = threadIdx.x;
    const int lane = t & 31;
    const int wave = t >> 5;
    const int lm   = lane & 15;
    const int hi   = lane >> 4;
    const int qb   = blockIdx.x;             // query block (64 rows)
    const int bh   = blockIdx.y;             // b*H + h
    const int q0   = qb * 64;

    const _Float16* Q = Qh + (size_t)bh * Sc * DKc;
    const _Float16* K = Kh + (size_t)bh * Sc * DKc;
    const _Float16* V = Vt + (size_t)bh * DKc * Sc;

    auto issue_kv = [&](int kb, int buf) {
#pragma unroll
        for (int i = 0; i < 4; ++i) {
            int u = i * 128 + t;
            int r = u >> 3, c8 = u & 7;
            async_copy_b128(&Ks[buf][r * LD + c8 * 8],
                            K + (size_t)(kb * 64 + r) * DKc + c8 * 8);
            async_copy_b128(&Vs[buf][r * LD + c8 * 8],
                            V + (size_t)r * Sc + kb * 64 + c8 * 8);
        }
    };

    // async-stage Q tile (64 x 64 halfs) and the first K/V block
#pragma unroll
    for (int i = 0; i < 4; ++i) {
        int u = i * 128 + t;
        int r = u >> 3, c8 = u & 7;
        async_copy_b128(&Qs[r * LD + c8 * 8],
                        Q + (size_t)(q0 + r) * DKc + c8 * 8);
    }
    issue_kv(0, 0);
    wait_async0();
    __syncthreads();

    // Q A-fragments live in registers for the whole kernel (K-dim 64 = 2 steps)
    AF qa[2];
    {
        int row = wave * 16 + lm;
#pragma unroll
        for (int s = 0; s < 2; ++s) {
            qa[s].h[0] = *reinterpret_cast<const h8*>(&Qs[row * LD + s * 32 + hi * 8]);
            qa[s].h[1] = *reinterpret_cast<const h8*>(&Qs[row * LD + s * 32 + 16 + hi * 8]);
        }
    }

    v8f Ov[4];
    float mst[8], lst[8];
#pragma unroll
    for (int nt = 0; nt < 4; ++nt) Ov[nt] = zero8();
#pragma unroll
    for (int r = 0; r < 8; ++r) { mst[r] = -3.0e38f; lst[r] = 0.0f; }

    const float scale = 0.125f;              // 1/sqrt(64)
    const int qw0 = q0 + wave * 16;

    for (int kb = 0; kb <= qb; ++kb) {
        const int cur = kb & 1;
        if (kb > 0) {
            wait_async0();                   // buffer `cur` has arrived
            __syncthreads();                 // everyone done reading buf cur^1
        }
        if (kb < qb) issue_kv(kb + 1, cur ^ 1);   // DMA overlaps compute below

        const _Float16* Kt = &Ks[cur][0];
        const _Float16* Vv = &Vs[cur][0];

        // ---- scores S = Q K^T (16 x 64 per wave) ----
        v8f sacc[4];
#pragma unroll
        for (int nt = 0; nt < 4; ++nt) {
            sacc[nt] = zero8();
            int row = nt * 16 + lm;
            AF b0, b1;
            b0.h[0] = *reinterpret_cast<const h8*>(&Kt[row * LD + hi * 8]);
            b0.h[1] = *reinterpret_cast<const h8*>(&Kt[row * LD + 16 + hi * 8]);
            b1.h[0] = *reinterpret_cast<const h8*>(&Kt[row * LD + 32 + hi * 8]);
            b1.h[1] = *reinterpret_cast<const h8*>(&Kt[row * LD + 32 + 16 + hi * 8]);
            sacc[nt] = WMMA_F16(qa[0].v, b0.v, sacc[nt]);
            sacc[nt] = WMMA_F16(qa[1].v, b1.v, sacc[nt]);
        }

        // ---- scale + causal mask (reference: scale first, then -1e9 fill) ----
#pragma unroll
        for (int nt = 0; nt < 4; ++nt) {
#pragma unroll
            for (int r = 0; r < 8; ++r) {
                float v = sacc[nt][r] * scale;
                if (kb == qb) {
                    int gq = qw0 + r + hi * 8;
                    int gk = kb * 64 + nt * 16 + lm;
                    if (gk > gq) v = -1e9f;
                }
                sacc[nt][r] = v;
            }
        }

        // ---- online softmax (row m = r + 8*hi spans 16 lanes of one half) ----
#pragma unroll
        for (int r = 0; r < 8; ++r) {
            float rmax = fmaxf(fmaxf(sacc[0][r], sacc[1][r]),
                               fmaxf(sacc[2][r], sacc[3][r]));
#pragma unroll
            for (int m = 1; m < 16; m <<= 1)
                rmax = fmaxf(rmax, __shfl_xor(rmax, m, 32));
            float mnew  = fmaxf(mst[r], rmax);
            float alpha = __expf(mst[r] - mnew);
            float rsum  = 0.0f;
#pragma unroll
            for (int nt = 0; nt < 4; ++nt) {
                float p = __expf(sacc[nt][r] - mnew);
                sacc[nt][r] = p;
                rsum += p;
            }
#pragma unroll
            for (int m = 1; m < 16; m <<= 1)
                rsum += __shfl_xor(rsum, m, 32);
            lst[r] = lst[r] * alpha + rsum;
            mst[r] = mnew;
#pragma unroll
            for (int nt = 0; nt < 4; ++nt) Ov[nt][r] *= alpha;
        }

        // ---- P: C-layout -> A-layout via per-wave LDS region ----
#pragma unroll
        for (int nt = 0; nt < 4; ++nt)
#pragma unroll
            for (int r = 0; r < 8; ++r)
                Ps[(wave * 16 + r + hi * 8) * LD + nt * 16 + lm] =
                    (_Float16)sacc[nt][r];

        AF pa[2];
        {
            int row = wave * 16 + lm;
#pragma unroll
            for (int s = 0; s < 2; ++s) {
                pa[s].h[0] = *reinterpret_cast<const h8*>(&Ps[row * LD + s * 32 + hi * 8]);
                pa[s].h[1] = *reinterpret_cast<const h8*>(&Ps[row * LD + s * 32 + 16 + hi * 8]);
            }
        }

        // ---- O += P V  (V^T rows are key-contiguous -> clean B-fragments) ----
#pragma unroll
        for (int nt = 0; nt < 4; ++nt) {
            int row = nt * 16 + lm;
            AF b0, b1;
            b0.h[0] = *reinterpret_cast<const h8*>(&Vv[row * LD + hi * 8]);
            b0.h[1] = *reinterpret_cast<const h8*>(&Vv[row * LD + 16 + hi * 8]);
            b1.h[0] = *reinterpret_cast<const h8*>(&Vv[row * LD + 32 + hi * 8]);
            b1.h[1] = *reinterpret_cast<const h8*>(&Vv[row * LD + 32 + 16 + hi * 8]);
            Ov[nt] = WMMA_F16(pa[0].v, b0.v, Ov[nt]);
            Ov[nt] = WMMA_F16(pa[1].v, b1.v, Ov[nt]);
        }
    }

    // ---- normalize + store back to [B, S, H*DK] f16 for the output GEMM ----
    const int b = bh >> 4;
    const int h = bh & (Hc - 1);
#pragma unroll
    for (int nt = 0; nt < 4; ++nt) {
#pragma unroll
        for (int r = 0; r < 8; ++r) {
            int gq  = q0 + wave * 16 + r + hi * 8;
            int col = h * DKc + nt * 16 + lm;
            attnOut[((size_t)(b * Sc + gq)) * Dc + col] =
                (_Float16)(Ov[nt][r] / lst[r]);
        }
    }
}

// ---------------------------------------------------------------------------
extern "C" void kernel_launch(void* const* d_in, const int* in_sizes, int n_in,
                              void* d_out, int out_size, void* d_ws, size_t ws_size,
                              hipStream_t stream) {
    (void)in_sizes; (void)n_in; (void)out_size; (void)ws_size;

    const float* q  = (const float*)d_in[0];
    const float* k  = (const float*)d_in[1];
    const float* v  = (const float*)d_in[2];
    // d_in[3] = attn_mask (exact causal tril) -> computed analytically
    const float* Wq = (const float*)d_in[4];
    const float* bq = (const float*)d_in[5];
    const float* Wk = (const float*)d_in[6];
    const float* bk = (const float*)d_in[7];
    const float* Wv = (const float*)d_in[8];
    const float* bv = (const float*)d_in[9];
    const float* Wo = (const float*)d_in[10];
    const float* bo = (const float*)d_in[11];
    float* out = (float*)d_out;

    const size_t headElems = (size_t)Bc * Hc * Sc * DKc;   // 4M halfs = 8 MB
    _Float16* Qh = (_Float16*)d_ws;
    _Float16* Kh = Qh + headElems;
    _Float16* Vt = Kh + headElems;
    _Float16* At = Vt + headElems;                          // [NTOK, D] f16

    dim3 gGemm(NTOK / 128, Dc / 64);
    proj_gemm_kernel<0><<<gGemm, 256, 0, stream>>>(q, nullptr, Wq, bq, Qh, nullptr);
    proj_gemm_kernel<0><<<gGemm, 256, 0, stream>>>(k, nullptr, Wk, bk, Kh, nullptr);
    proj_gemm_kernel<1><<<gGemm, 256, 0, stream>>>(v, nullptr, Wv, bv, Vt, nullptr);

    flash_attn_kernel<<<dim3(Sc / 64, Bc * Hc), 128, 0, stream>>>(Qh, Kh, Vt, At);

    proj_gemm_kernel<2><<<gGemm, 256, 0, stream>>>(nullptr, At, Wo, bo, nullptr, out);
}
